// EncoderEstimator_10883447128727
// MI455X (gfx1250) — compile-verified, matrix operand order checked by reading, not run on hardware
//
#include <hip/hip_runtime.h>
#include <hip/hip_bf16.h>

typedef __attribute__((ext_vector_type(16))) _Float16 v16h;
typedef __attribute__((ext_vector_type(8)))  float    v8f;
typedef __attribute__((ext_vector_type(4)))  float    f4v;

#define Hh 64
#define Ww 64
#define HW 4096
#define Bb 64
#define Nn 17
#define Cc 32
#define BN (Bb*Nn)

// ---------- closed form of the 15-move ring weight map ----------
// A border row/col is written exactly once, at m = distance-to-border
// (or m=1 if the keypoint sits on that border); interior matches at
// m = |delta| along that axis, gated by the perpendicular |delta| <= m.
// Final value = largest matching m (later moves overwrite earlier ones).
__device__ __forceinline__ float ring_weight(int x, int y, int kx, int ky) {
    int dx = x - kx, dy = y - ky;
    int adx = dx < 0 ? -dx : dx, ady = dy < 0 ? -dy : dy;
    int best = 0;
    int mL = (x > 0)  ? ((dx < 0) ? -dx : 0) : max(kx, 1);
    if (mL >= 1 && mL <= 15 && ady <= mL && mL > best) best = mL;
    int mR = (x < 63) ? ((dx > 0) ?  dx : 0) : max(63 - kx, 1);
    if (mR >= 1 && mR <= 15 && ady <= mR && mR > best) best = mR;
    int mU = (y > 0)  ? ((dy < 0) ? -dy : 0) : max(ky, 1);
    if (mU >= 1 && mU <= 15 && adx <= mU && mU > best) best = mU;
    int mD = (y < 63) ? ((dy > 0) ?  dy : 0) : max(63 - ky, 1);
    if (mD >= 1 && mD <= 15 && adx <= mD && mD > best) best = mD;
    return best ? (0.5f + 0.25f * (float)best) : 10.0f;
}

// ---------- K1: WMMA dot + dist + weight -> ws.wdist ----------
// grid = (64 batches, 32 pixel-tile groups), block = 256 (8 waves).
// Each wave owns one 16-pixel tile: two v_wmma_f32_16x16x32_f16
// (n-tiles 0..15 and 16..31, rows >=17 zero-padded). C=32 == K in one shot.
__global__ void k1_wmma_dist(const float* __restrict__ kq,
                             const float* __restrict__ keys,
                             const int*   __restrict__ joints,
                             float* __restrict__ wdist) {
    __shared__ float sk[32 * 32];   // keys f32, zero padded to 32 rows
    __shared__ float k2s[32];
    __shared__ int   sjx[32], sjy[32];

    const int b    = blockIdx.x;
    const int lane = threadIdx.x & 31;
    const int wave = threadIdx.x >> 5;
    const int half = lane >> 4;
    const int col  = lane & 15;

    for (int i = threadIdx.x; i < 1024; i += 256) {
        int n = i >> 5, c = i & 31;
        sk[i] = (n < Nn) ? keys[(b * Nn + n) * Cc + c] : 0.0f;
    }
    __syncthreads();
    if (threadIdx.x < 32) {
        float s = 0.0f;
        #pragma unroll
        for (int c = 0; c < 32; ++c) { float v = sk[threadIdx.x * 32 + c]; s += v * v; }
        k2s[threadIdx.x] = s;
    }
    if (threadIdx.x >= 32 && threadIdx.x < 32 + Nn) {
        int n = threadIdx.x - 32;
        sjx[n] = joints[(b * Nn + n) * 2 + 0];
        sjy[n] = joints[(b * Nn + n) * 2 + 1];
    }
    __syncthreads();

    // A-matrix VGPR layout (16-bit A 16x32): lane holds row M=lane&15,
    // half i<8 -> K = i + 8*half ; i>=8 -> K = i + 8 + 8*half.
    v16h a0, a1;
    #pragma unroll
    for (int i = 0; i < 16; ++i) {
        int k = i + 8 * half + ((i >= 8) ? 8 : 0);
        a0[i] = (_Float16)sk[col * 32 + k];
        a1[i] = (_Float16)sk[(16 + col) * 32 + k];   // only row 16 (n=16) nonzero
    }

    const int t = blockIdx.y * 8 + wave;        // pixel tile 0..255
    const int p = t * 16 + col;                 // this lane's pixel (B column / D column)
    const float* __restrict__ src = kq + (size_t)b * Cc * HW + p;

    // B-matrix (32x16): lane holds column = pixel, same K split per half.
    // Accumulate partial ||q||^2 for the 16 channels this lane converts.
    v16h bm; float q2p = 0.0f;
    #pragma unroll
    for (int i = 0; i < 16; ++i) {
        int k = i + 8 * half + ((i >= 8) ? 8 : 0);
        float v = src[(size_t)k * HW];
        q2p += v * v;
        bm[i] = (_Float16)v;
    }
    float q2 = q2p + __shfl_xor(q2p, 16);       // lanes L and L^16 share a pixel

    v8f c0 = {}, c1 = {};
    c0 = __builtin_amdgcn_wmma_f32_16x16x32_f16(false, a0, false, bm, (short)0, c0, false, false);
    c1 = __builtin_amdgcn_wmma_f32_16x16x32_f16(false, a1, false, bm, (short)0, c1, false, false);

    const int x = p & 63, y = p >> 6;
    #pragma unroll
    for (int v = 0; v < 8; ++v) {
        int n0 = v + 8 * half;                  // D row M = v + 8*(lane>=16)
        float d2 = q2 + k2s[n0] - 2.0f * c0[v];
        float dist = sqrtf(fmaxf(d2, 0.0f));
        float w = ring_weight(x, y, sjx[n0], sjy[n0]);
        wdist[(size_t)(b * Nn + n0) * HW + p] = dist * w;
        if (v == 0 && half == 0) {              // second tile: only M=0 -> n=16
            float e2 = q2 + k2s[16] - 2.0f * c1[0];
            float ed = sqrtf(fmaxf(e2, 0.0f));
            float ew = ring_weight(x, y, sjx[16], sjy[16]);
            wdist[(size_t)(b * Nn + 16) * HW + p] = ed * ew;
        }
    }
}

// ---------- K2: per-(b,n) min/argmin over 4096 (packed u64, float4 reads) ----------
__global__ void k2_reduce_hw(const float* __restrict__ wdist,
                             unsigned long long* __restrict__ min_hw) {
    __shared__ unsigned long long red[128];
    const int bn = blockIdx.x, tid = threadIdx.x;
    const f4v* __restrict__ src4 = (const f4v*)(wdist + (size_t)bn * HW);
    unsigned long long key = ~0ull;
    #pragma unroll
    for (int j = 0; j < 8; ++j) {               // 1024 float4 / 128 threads
        int q = tid + j * 128;
        if (j + 1 < 8) __builtin_prefetch(&src4[q + 128], 0, 1);
        f4v v = src4[q];
        #pragma unroll
        for (int e = 0; e < 4; ++e) {
            unsigned long long k =
                ((unsigned long long)__float_as_uint(v[e]) << 32) | (unsigned)(q * 4 + e);
            key = k < key ? k : key;
        }
    }
    red[tid] = key; __syncthreads();
    for (int s = 64; s > 0; s >>= 1) {
        if (tid < s) { unsigned long long o = red[tid + s]; if (o < red[tid]) red[tid] = o; }
        __syncthreads();
    }
    if (tid == 0) min_hw[bn] = red[0];
}

// ---------- K3: 4x bilinear upsample + argmin ----------
// One workgroup per (b,n). Stage the 16 KB source tile in LDS (read once from
// L2), then each thread emits float4 chunks with non-temporal b128 stores.
// 4x upsample is separable with fixed phase weights {.625,.875,.125,.375}:
// outputs 4k..4k+3 need only source columns k-1,k,k+1.
__global__ void k3_upsample(const float* __restrict__ wdist,
                            float* __restrict__ out_x4,
                            unsigned long long* __restrict__ min_x4) {
    __shared__ float s[HW];                       // 64x64 source tile (16 KB)
    __shared__ unsigned long long red[256];
    const int bn = blockIdx.x, tid = threadIdx.x;
    const f4v* __restrict__ src4 = (const f4v*)(wdist + (size_t)bn * HW);
    f4v* __restrict__ dst4 = (f4v*)(out_x4 + (size_t)bn * (256 * 256));

    #pragma unroll
    for (int i = 0; i < 4; ++i)                   // 1024 float4 / 256 threads
        ((f4v*)s)[tid + i * 256] = src4[tid + i * 256];
    __syncthreads();

    unsigned long long key = ~0ull;
    for (int c = tid; c < 16384; c += 256) {      // 16384 float4 chunks
        int oy = c >> 6;                          // output row 0..255
        int cx = c & 63;                          // chunk: outputs ox = 4cx..4cx+3
        int y0 = ((oy + 2) >> 2) - 1;             // floor((oy+0.5)/4 - 0.5)
        float wy = 0.125f + 0.25f * (float)((oy + 2) & 3);  // {.625,.875,.125,.375}
        int y0c = max(y0, 0), y1c = min(y0 + 1, 63);
        const float* r0 = s + y0c * Ww;
        const float* r1 = s + y1c * Ww;
        int xm = max(cx - 1, 0), xp = min(cx + 1, 63);
        float a0 = r0[xm], a1 = r0[cx], a2 = r0[xp];
        float b0 = r1[xm], b1 = r1[cx], b2 = r1[xp];
        float m0 = a0 + wy * (b0 - a0);           // vertical blend (shared by 4 outputs)
        float m1 = a1 + wy * (b1 - a1);
        float m2 = a2 + wy * (b2 - a2);
        f4v o;
        o[0] = m0 + 0.625f * (m1 - m0);           // wx for ox%4 = 0
        o[1] = m0 + 0.875f * (m1 - m0);           // 1
        o[2] = m1 + 0.125f * (m2 - m1);           // 2
        o[3] = m1 + 0.375f * (m2 - m1);           // 3
        __builtin_nontemporal_store(o, &dst4[c]); // 285 MB write stream: bypass cache
        int base = oy * 256 + cx * 4;
        #pragma unroll
        for (int e = 0; e < 4; ++e) {
            unsigned long long k =
                ((unsigned long long)__float_as_uint(o[e]) << 32) | (unsigned)(base + e);
            key = k < key ? k : key;
        }
    }
    red[tid] = key; __syncthreads();
    for (int sdx = 128; sdx > 0; sdx >>= 1) {
        if (tid < sdx) { unsigned long long o = red[tid + sdx]; if (o < red[tid]) red[tid] = o; }
        __syncthreads();
    }
    if (tid == 0) min_x4[bn] = red[0];
}

// ---------- K4: finalize joints / valid / new_keys (1 wave per (b,n)) ----------
__global__ void k4_finalize(const float* __restrict__ kq,
                            const float* __restrict__ keys,
                            const unsigned long long* __restrict__ min_hw,
                            const unsigned long long* __restrict__ min_x4,
                            float* __restrict__ out) {
    const int bn = blockIdx.x;
    const int b = bn / Nn;
    const int lane = threadIdx.x;           // 32 lanes == C channels
    int*   out_joints = (int*)out;                      // [0, 2176)
    int*   out_valid  = (int*)out + Bb * Nn * 2;        // [2176, 3264)
    float* out_nk     = out + Bb * Nn * 3;              // [3264, 38080)

    float kv = keys[(size_t)bn * Cc + lane];
    bool nz = (__ballot(kv != 0.0f) != 0ull);

    unsigned long long mh = min_hw[bn];
    float mind = __uint_as_float((unsigned)(mh >> 32));
    int   idx  = (int)(mh & 0xffffffffu);
    int valid = (nz && ((int)floorf(mind) <= 5)) ? 1 : 0;

    float g = kq[(size_t)b * Cc * HW + (size_t)lane * HW + idx];
    out_nk[(size_t)bn * Cc + lane] = valid ? g : kv;

    if (lane == 0) {
        out_valid[bn] = valid;
        unsigned long long mx = min_x4[bn];
        int sidx = (int)(mx & 0xffffffffu);
        out_joints[bn * 2 + 0] = valid ? (sidx >> 8)  : -1;   // sv
        out_joints[bn * 2 + 1] = valid ? (sidx & 255) : -1;   // sh
    }
}

extern "C" void kernel_launch(void* const* d_in, const int* in_sizes, int n_in,
                              void* d_out, int out_size, void* d_ws, size_t ws_size,
                              hipStream_t stream) {
    const float* kq     = (const float*)d_in[0];   // (64,32,64,64) f32
    const float* keys   = (const float*)d_in[1];   // (64,17,32)   f32
    const int*   joints = (const int*)  d_in[2];   // (64,17,2)    i32
    float* out = (float*)d_out;

    // workspace: wdist (17.8 MB, stays L2-resident) + two 1088-entry u64 tables
    float* wdist = (float*)d_ws;
    unsigned long long* min_hw =
        (unsigned long long*)((char*)d_ws + (size_t)BN * HW * sizeof(float));
    unsigned long long* min_x4 = min_hw + BN;

    float* out_x4 = out + (size_t)BN * 2 + BN + (size_t)BN * Cc;  // offset 38080

    dim3 g1(Bb, 32);
    k1_wmma_dist<<<g1, 256, 0, stream>>>(kq, keys, joints, wdist);
    k2_reduce_hw<<<BN, 128, 0, stream>>>(wdist, min_hw);
    k3_upsample <<<BN, 256, 0, stream>>>(wdist, out_x4, min_x4);
    k4_finalize <<<BN,  32, 0, stream>>>(kq, keys, min_hw, min_x4, out);
}
